// EmbeddingAlignmentGNN_24352464570114
// MI455X (gfx1250) — matrix operand order repaired and checked by reading, not compiled, not run on hardware
//
#include <hip/hip_runtime.h>
#include <hip/hip_bf16.h>

// ---------------------------------------------------------------------------
// Problem constants (match reference)
// ---------------------------------------------------------------------------
#define N_E 100000
#define N_C 50000
#define HDIM 256
#define D_E 768
#define D_C 1024

typedef __attribute__((ext_vector_type(16))) __bf16 v16bf;
typedef __attribute__((ext_vector_type(8)))  float  v8f;

union Frag16 { unsigned int u[8]; v16bf v; };

__device__ __forceinline__ unsigned short f2bf(float f) {
    unsigned int x = __float_as_uint(f);
    x += 0x7fffu + ((x >> 16) & 1u);   // round-to-nearest-even
    return (unsigned short)(x >> 16);
}

// ---------------------------------------------------------------------------
// Weight prep: f32 [K][256] -> bf16 transposed [256][K] (tiny, once per call)
// ---------------------------------------------------------------------------
__global__ __launch_bounds__(256)
void cvt_transpose_bf16(const float* __restrict__ src, unsigned short* __restrict__ dst,
                        int K) {
    const long long i = (long long)blockIdx.x * blockDim.x + threadIdx.x;
    if (i >= (long long)K * HDIM) return;
    const int k = (int)(i >> 8);
    const int n = (int)(i & 255);
    dst[(size_t)n * K + k] = f2bf(src[i]);
}

// ---------------------------------------------------------------------------
// Fused multi-term GEMM:
//   C[M x 256] = sum_t ( rowscale_t(row) * A_t[M x K_t] ) @ B_t[K_t x 256]
//                + sum_t bias_t
// A f32 (scaled + converted to bf16 while staged into LDS); B pre-converted
// bf16, pre-transposed to [256][K_t], staged with ASYNC global->LDS loads.
// Block = 256 threads = 8 waves. Tile = 64 rows x 256 cols as 4 strips of 16
// sharing one staged B tile -> 8 WMMAs/wave/k-step. C written exactly once.
// Requires M % 16 == 0, K % 32 == 0. Partial 64-row blocks handled per strip.
// ---------------------------------------------------------------------------
struct Term {
    const float* A;               // [M][K]
    const float* rowscale;        // optional count vector -> scale 1/max(c,1)
    const unsigned short* Bt;     // bf16 [256][K]
    const float* bias;            // optional [256]
    int K;
};
struct GemmParams { Term t[4]; int nterms; };

__global__ __launch_bounds__(256)
void sage_gemm_wmma(GemmParams P, float* __restrict__ C, int M) {
    __shared__ __attribute__((aligned(16))) unsigned short la[64 * 32];    // [m][k], 4 KB
    __shared__ __attribute__((aligned(16))) unsigned short lb[256 * 32];   // [n][k], 16 KB

    const int t    = threadIdx.x;
    const int lane = t & 31;
    const int wave = t >> 5;
    const int m0   = blockIdx.x * 64;
    const int colbase = wave * 32;           // each wave owns 32 output columns

    v8f acc[4][2];
#pragma unroll
    for (int s = 0; s < 4; ++s) { acc[s][0] = {}; acc[s][1] = {}; }

    // A staging assignment: 2048 elems, 8 consecutive per thread
    const int ar = (t * 8) >> 5;             // LDS row 0..63
    const int ak = (t * 8) & 31;             // k offset {0,8,16,24}
    const bool arow_ok = (m0 + ar) < M;

    // LDS byte offset of this thread's B column slice (flat LDS aperture:
    // low 32 bits are the LDS offset, ISA 10.2)
    const unsigned lb_off = (unsigned)(unsigned long long)(&lb[t * 32]);

    for (int term = 0; term < P.nterms; ++term) {
        const Term tm = P.t[term];
        const int K = tm.K;
        float rs = 1.0f;
        if (tm.rowscale && arow_ok) rs = 1.0f / fmaxf(tm.rowscale[m0 + ar], 1.0f);
        const float* aptr = tm.A + (size_t)(m0 + ar) * K + ak;
        const unsigned short* btp = tm.Bt + (size_t)t * K;   // thread t = column n
        const int nsteps = K >> 5;

        for (int s = 0; s < nsteps; ++s) {
            const int k0 = s << 5;
            // --- stage B tile: async copy 64 contiguous bytes global -> LDS
            {
                const unsigned long long ga = (unsigned long long)(btp + k0);
                asm volatile(
                    "global_load_async_to_lds_b128 %0, %1, off\n\t"
                    "global_load_async_to_lds_b128 %0, %1, off offset:16\n\t"
                    "global_load_async_to_lds_b128 %0, %1, off offset:32\n\t"
                    "global_load_async_to_lds_b128 %0, %1, off offset:48"
                    :: "v"(lb_off), "v"(ga) : "memory");
            }
            // --- stage A tile 64x32 (scale + convert), zeros for OOB strips
            {
                unsigned short* lr = &la[ar * 32 + ak];
                if (arow_ok) {
                    const float* ap = aptr + k0;
#pragma unroll
                    for (int j = 0; j < 8; ++j) lr[j] = f2bf(ap[j] * rs);
                    if (s + 1 < nsteps) __builtin_prefetch(ap + 32, 0, 1);
                } else {
#pragma unroll
                    for (int j = 0; j < 8; ++j) lr[j] = 0;
                }
            }
            asm volatile("s_wait_asynccnt 0" ::: "memory");
            __syncthreads();

            // --- build B fragments once (lane = column n, halves K = 16*g + h)
            Frag16 fb0, fb1;
            {
                const int nl = lane & 15, g = lane >> 4;
                const unsigned short* b0 = &lb[(colbase + nl) * 32 + 16 * g];
                const unsigned short* b1 = &lb[(colbase + 16 + nl) * 32 + 16 * g];
#pragma unroll
                for (int j = 0; j < 8; ++j) {
                    fb0.u[j] = *(const unsigned int*)&b0[2 * j];
                    fb1.u[j] = *(const unsigned int*)&b1[2 * j];
                }
            }
            // --- 4 strips x 2 WMMAs, sharing the staged B tile
            const int m = lane & 15, g = lane >> 4;
#pragma unroll
            for (int st = 0; st < 4; ++st) {
                Frag16 fa;
                const unsigned short* arow = &la[(st * 16 + m) * 32];
#pragma unroll
                for (int i = 0; i < 8; ++i) {
                    const int kk = 16 * g + 2 * i;
                    fa.u[i] = *(const unsigned int*)&arow[kk];
                }
                acc[st][0] = __builtin_amdgcn_wmma_f32_16x16x32_bf16(
                    false, fa.v, false, fb0.v, (short)0, acc[st][0], false, false);
                acc[st][1] = __builtin_amdgcn_wmma_f32_16x16x32_bf16(
                    false, fa.v, false, fb1.v, (short)0, acc[st][1], false, false);
            }
            __syncthreads();
        }
    }

    // --- writeback (f32 C/D 16x16 layout: VGPR r -> M = r + 8*g, lane -> N)
    const int nl = lane & 15, g = lane >> 4;
    const int c0 = colbase + nl;
    const int c1 = colbase + 16 + nl;
    float b0v = 0.0f, b1v = 0.0f;
    for (int term = 0; term < P.nterms; ++term) {
        if (P.t[term].bias) { b0v += P.t[term].bias[c0]; b1v += P.t[term].bias[c1]; }
    }
#pragma unroll
    for (int st = 0; st < 4; ++st) {
        if (m0 + st * 16 >= M) break;        // block-uniform strip guard
#pragma unroll
        for (int r = 0; r < 8; ++r) {
            const int m = m0 + st * 16 + 8 * g + r;
            C[(size_t)m * HDIM + c0] = acc[st][0][r] + b0v;
            C[(size_t)m * HDIM + c1] = acc[st][1][r] + b1v;
        }
    }
}

// ---------------------------------------------------------------------------
// Scatter-add of H=256 features per edge + degree count. 64 threads/edge,
// float4 gather, 4 f32 atomics per thread (dst agg is L2-resident).
// ---------------------------------------------------------------------------
__global__ __launch_bounds__(256)
void scatter_add_edges(const float* __restrict__ feat, const int* __restrict__ esrc,
                       const int* __restrict__ edst, int E,
                       float* __restrict__ agg, float* __restrict__ cnt) {
    const long long idx = (long long)blockIdx.x * blockDim.x + threadIdx.x;
    const long long e = idx >> 6;
    if (e >= E) return;
    const int f = (int)(idx & 63) * 4;
    const int s = esrc[e];
    const int d = edst[e];
    const float4 v = *(const float4*)(feat + (size_t)s * HDIM + f);
    float* dp = agg + (size_t)d * HDIM + f;
    atomicAdd(dp + 0, v.x);
    atomicAdd(dp + 1, v.y);
    atomicAdd(dp + 2, v.z);
    atomicAdd(dp + 3, v.w);
    if (f == 0) atomicAdd(cnt + d, 1.0f);
}

__global__ __launch_bounds__(256)
void relu_residual(float* __restrict__ h, const float* __restrict__ o, long long n) {
    const long long i = (long long)blockIdx.x * blockDim.x + threadIdx.x;
    if (i < n) h[i] = fmaxf(o[i], 0.0f) + h[i];
}

// One wave per row: L2-normalize 256-wide rows into out.
__global__ __launch_bounds__(256)
void l2norm_rows(const float* __restrict__ o, float* __restrict__ out, int nrows) {
    const int wid  = blockIdx.x * (blockDim.x >> 5) + (threadIdx.x >> 5);
    const int lane = threadIdx.x & 31;
    if (wid >= nrows) return;
    const float* row = o + (size_t)wid * HDIM;
    float v[8];
    float s = 0.0f;
#pragma unroll
    for (int j = 0; j < 8; ++j) {
        v[j] = row[lane + j * 32];
        s += v[j] * v[j];
    }
#pragma unroll
    for (int m = 16; m > 0; m >>= 1) s += __shfl_xor(s, m, 32);
    const float inv = 1.0f / fmaxf(sqrtf(s), 1e-12f);
    float* orow = out + (size_t)wid * HDIM;
#pragma unroll
    for (int j = 0; j < 8; ++j) orow[lane + j * 32] = v[j] * inv;
}

// ---------------------------------------------------------------------------
// Orchestration
// ---------------------------------------------------------------------------
extern "C" void kernel_launch(void* const* d_in, const int* in_sizes, int n_in,
                              void* d_out, int out_size, void* d_ws, size_t ws_size,
                              hipStream_t stream) {
    const float* x_e  = (const float*)d_in[0];
    const float* x_c  = (const float*)d_in[1];
    const int*   e_ee = (const int*)d_in[2];
    const int*   e_ec = (const int*)d_in[3];
    const int*   e_ce = (const int*)d_in[4];
    const float* P_e  = (const float*)d_in[5];
    const float* P_c  = (const float*)d_in[6];
    const float* Wl1  = (const float*)d_in[7];
    const float* bl1  = (const float*)d_in[8];
    const float* Wr1  = (const float*)d_in[9];
    const float* Wl2  = (const float*)d_in[10];
    const float* bl2  = (const float*)d_in[11];
    const float* Wr2  = (const float*)d_in[12];

    const int E_ee = in_sizes[2] / 2;
    const int E_ec = in_sizes[3] / 2;
    const int E_ce = in_sizes[4] / 2;

    const size_t SE = (size_t)N_E * HDIM * sizeof(float);   // 102.4 MB
    const size_t SC = (size_t)N_C * HDIM * sizeof(float);   //  51.2 MB
    const size_t HH = (size_t)HDIM * HDIM;

    char* p = (char*)d_ws;
    float* h_e     = (float*)p; p += SE;
    float* h_c     = (float*)p; p += SC;
    float* mean_ee = (float*)p; p += SE;   // raw sums; divided during GEMM staging
    float* mean_ce = (float*)p; p += SE;
    float* mean_ec = (float*)p; p += SC;
    float* o_e     = (float*)p; p += SE;
    float* o_c     = (float*)p; p += SC;
    float* cnt_ee  = (float*)p; p += (size_t)N_E * sizeof(float);
    float* cnt_ce  = (float*)p; p += (size_t)N_E * sizeof(float);
    float* cnt_ec  = (float*)p; p += (size_t)N_C * sizeof(float);
    // bf16 transposed weight copies [256][K]
    unsigned short* wPeT  = (unsigned short*)p; p += (size_t)D_E * HDIM * 2;
    unsigned short* wPcT  = (unsigned short*)p; p += (size_t)D_C * HDIM * 2;
    unsigned short* wWl1T = (unsigned short*)p; p += 3 * HH * 2;
    unsigned short* wWr1T = (unsigned short*)p; p += 3 * HH * 2;
    unsigned short* wWl2T = (unsigned short*)p; p += 3 * HH * 2;
    unsigned short* wWr2T = (unsigned short*)p; p += 3 * HH * 2;

    const dim3 blk(256);
    auto cvtT = [&](const float* s, unsigned short* d, int K) {
        const long long n = (long long)K * HDIM;
        cvt_transpose_bf16<<<dim3((unsigned)((n + 255) / 256)), blk, 0, stream>>>(s, d, K);
    };
    auto gemm = [&](const GemmParams& P, float* C, int M) {
        sage_gemm_wmma<<<dim3((M + 63) / 64), blk, 0, stream>>>(P, C, M);
    };
    auto scatter = [&](const float* feat, const int* edge, int E, float* agg, float* cnt) {
        const long long thr = (long long)E * 64;
        scatter_add_edges<<<dim3((unsigned)((thr + 255) / 256)), blk, 0, stream>>>(
            feat, edge, edge + E, E, agg, cnt);
    };

    // ---- prep: bf16 + transpose all weights ---------------------------------
    cvtT(P_e, wPeT, D_E);
    cvtT(P_c, wPcT, D_C);
    for (int s = 0; s < 3; ++s) {
        cvtT(Wl1 + s * HH, wWl1T + s * HH, HDIM);
        cvtT(Wr1 + s * HH, wWr1T + s * HH, HDIM);
        cvtT(Wl2 + s * HH, wWl2T + s * HH, HDIM);
        cvtT(Wr2 + s * HH, wWr2T + s * HH, HDIM);
    }

    // ---- per-type projections: h = x @ P ------------------------------------
    { GemmParams P{{{x_e, nullptr, wPeT, nullptr, D_E}}, 1}; gemm(P, h_e, N_E); }
    { GemmParams P{{{x_c, nullptr, wPcT, nullptr, D_C}}, 1}; gemm(P, h_c, N_C); }

    for (int layer = 0; layer < 2; ++layer) {
        const unsigned short* WlT = layer ? wWl2T : wWl1T;
        const unsigned short* WrT = layer ? wWr2T : wWr1T;
        const float* bl = layer ? bl2 : bl1;

        // ---- aggregation (raw sums + degree counts) -------------------------
        hipMemsetAsync(mean_ee, 0, SE, stream);
        hipMemsetAsync(cnt_ee, 0, (size_t)N_E * sizeof(float), stream);
        scatter(h_e, e_ee, E_ee, mean_ee, cnt_ee);

        hipMemsetAsync(mean_ce, 0, SE, stream);
        hipMemsetAsync(cnt_ce, 0, (size_t)N_E * sizeof(float), stream);
        scatter(h_c, e_ce, E_ce, mean_ce, cnt_ce);

        hipMemsetAsync(mean_ec, 0, SC, stream);
        hipMemsetAsync(cnt_ec, 0, (size_t)N_C * sizeof(float), stream);
        scatter(h_e, e_ec, E_ec, mean_ec, cnt_ec);

        // ---- fused linears: one C write per destination type ----------------
        {
            GemmParams P{{
                {mean_ee, cnt_ee, WlT + 0 * HH, bl + 0 * HDIM, HDIM},
                {h_e,     nullptr, WrT + 0 * HH, nullptr,      HDIM},
                {mean_ce, cnt_ce, WlT + 2 * HH, bl + 2 * HDIM, HDIM},
                {h_e,     nullptr, WrT + 2 * HH, nullptr,      HDIM}}, 4};
            gemm(P, o_e, N_E);
        }
        {
            GemmParams P{{
                {mean_ec, cnt_ec, WlT + 1 * HH, bl + 1 * HDIM, HDIM},
                {h_c,     nullptr, WrT + 1 * HH, nullptr,      HDIM}}, 2};
            gemm(P, o_c, N_C);
        }

        if (layer == 0) {
            // ---- relu + residual (in place) ---------------------------------
            const long long ne = (long long)N_E * HDIM, nc = (long long)N_C * HDIM;
            relu_residual<<<dim3((unsigned)((ne + 255) / 256)), blk, 0, stream>>>(h_e, o_e, ne);
            relu_residual<<<dim3((unsigned)((nc + 255) / 256)), blk, 0, stream>>>(h_c, o_c, nc);
        }
    }

    // ---- L2 normalize into output (o_e rows, then o_c rows) -----------------
    float* out = (float*)d_out;
    l2norm_rows<<<dim3((N_E + 7) / 8), blk, 0, stream>>>(o_e, out, N_E);
    l2norm_rows<<<dim3((N_C + 7) / 8), blk, 0, stream>>>(o_c, out + (size_t)N_E * HDIM, N_C);
}
